// LAME_78735340470307
// MI455X (gfx1250) — compile-verified, matrix operand order checked by reading, not run on hardware
//
#include <hip/hip_runtime.h>
#include <math.h>
#include <stdint.h>

// Problem constants (from reference)
constexpr int kN = 2048;
constexpr int kC = 1000;
constexpr int kD = 768;
constexpr int kKNN = 5;
constexpr float kLam = 1.0f;     // BOUND_LAMBDA
constexpr int kSteps = 100;      // MAX_STEPS
constexpr int kMaxDeg = 80;      // 5 fwd + up to 75 reverse edges (Poisson(5) tail safe)

typedef _Float16 v16h __attribute__((ext_vector_type(16)));
typedef _Float16 v8h  __attribute__((ext_vector_type(8)));
typedef float    v8f  __attribute__((ext_vector_type(8)));
typedef int      v4i  __attribute__((ext_vector_type(4)));

typedef __attribute__((address_space(1))) v4i* gv4i_ptr;   // global
typedef __attribute__((address_space(3))) v4i* lv4i_ptr;   // LDS

// ---- CDNA5 async global->LDS copy (16B per lane), ASYNCcnt-tracked ----------
__device__ __forceinline__ void async_cp16(const _Float16* g, _Float16* l) {
#if __has_builtin(__builtin_amdgcn_global_load_async_to_lds_b128)
    __builtin_amdgcn_global_load_async_to_lds_b128(
        (gv4i_ptr)g, (lv4i_ptr)l, 0, 0);
#else
    const unsigned loff =
        (unsigned)(uintptr_t)(__attribute__((address_space(3))) _Float16*)l;
    asm volatile("global_load_async_to_lds_b128 %0, %1, off"
                 :: "v"(loff), "v"(g) : "memory");
#endif
}

template <int Ncnt>
__device__ __forceinline__ void wait_async() {
#if __has_builtin(__builtin_amdgcn_s_wait_asynccnt)
    __builtin_amdgcn_s_wait_asynccnt((unsigned short)Ncnt);
#else
    if constexpr (Ncnt == 0) asm volatile("s_wait_asynccnt 0x0" ::: "memory");
    else                     asm volatile("s_wait_asynccnt 0x4" ::: "memory");
#endif
}

// ---------------------------------------------------------------- init flags
__global__ void lame_init(float* oldE, int* done) {
    if (threadIdx.x == 0) { *oldE = INFINITY; *done = 0; }
}

// ------------------------------------------------- row-normalize feats -> f16
__global__ void lame_normalize(const float* __restrict__ feats, _Float16* __restrict__ fh) {
    const int i = blockIdx.x, tid = threadIdx.x;
    __shared__ float red[256];
    float s = 0.f;
    for (int d = tid; d < kD; d += 256) { float v = feats[(size_t)i * kD + d]; s += v * v; }
    red[tid] = s; __syncthreads();
    for (int t = 128; t > 0; t >>= 1) { if (tid < t) red[tid] += red[tid + t]; __syncthreads(); }
    const float r = rsqrtf(red[0]);
    for (int d = tid; d < kD; d += 256)
        fh[(size_t)i * kD + d] = (_Float16)(feats[(size_t)i * kD + d] * r);
}

// ------------------------------------------------- Gram matrix via WMMA f16
// Block (8 waves) computes a 64x64 tile of sim = f_hat @ f_hat^T.
// A/B 64x64 k-chunks double-buffered in LDS via async global->LDS loads.
// Wave layout: 4 (M strips of 16) x 2 (N strips of 32); 48 WMMAs per wave.
__global__ void lame_gram_wmma(const _Float16* __restrict__ fh, float* __restrict__ sim) {
    constexpr int KC  = 64;          // k-chunk
    constexpr int NCH = kD / KC;     // 12 chunks
    __shared__ _Float16 As[2][64][KC];   // 2 x 8 KB
    __shared__ _Float16 Bs[2][64][KC];   // 2 x 8 KB

    const int tid  = threadIdx.x;
    const int lane = tid & 31;
    const int wid  = tid >> 5;           // 0..7
    const int wm   = wid >> 1;           // 0..3 -> 16-row strip
    const int wn   = wid & 1;            // 0..1 -> 32-col strip
    const int bm   = (blockIdx.x / (kN / 64)) * 64;
    const int bn   = (blockIdx.x % (kN / 64)) * 64;

    // Each thread copies 2 x 16B per side per chunk (64*64 halfs = 512 vec8).
    auto issue = [&](int buf, int ch) {
#pragma unroll
        for (int v = 0; v < 2; ++v) {
            const int idx = tid + 256 * v;
            const int row = idx >> 3;            // 8 vec8 per row
            const int kv  = (idx & 7) << 3;
            const int k0  = ch * KC;
            async_cp16(fh + (size_t)(bm + row) * kD + k0 + kv, &As[buf][row][kv]);
            async_cp16(fh + (size_t)(bn + row) * kD + k0 + kv, &Bs[buf][row][kv]);
        }
    };

    // Fragment addressing per ISA 16-bit A layout:
    //   lanes 0-15: M = lane&15, K pairs 0..7 / 16..23; lanes 16-31: K shifted +8
    const int fr   = lane & 15;
    const int koff = (lane >> 4) << 3;

    v8f acc0 = {}, acc1 = {};
    issue(0, 0);
    for (int ch = 0; ch < NCH; ++ch) {
        const int buf = ch & 1;
        if (ch + 1 < NCH) { issue(buf ^ 1, ch + 1); wait_async<4>(); }
        else              { wait_async<0>(); }
        __syncthreads();                         // chunk `ch` fully in LDS

#pragma unroll
        for (int ks = 0; ks < KC / 32; ++ks) {   // 2 k-steps of 32
            const int kb = ks * 32 + koff;
            v16h A, B0, B1;
            const v8h a0  = *(const v8h*)&As[buf][wm * 16 + fr][kb];
            const v8h a1  = *(const v8h*)&As[buf][wm * 16 + fr][kb + 16];
            const v8h b00 = *(const v8h*)&Bs[buf][wn * 32 + fr][kb];
            const v8h b01 = *(const v8h*)&Bs[buf][wn * 32 + fr][kb + 16];
            const v8h b10 = *(const v8h*)&Bs[buf][wn * 32 + 16 + fr][kb];
            const v8h b11 = *(const v8h*)&Bs[buf][wn * 32 + 16 + fr][kb + 16];
#pragma unroll
            for (int e = 0; e < 8; ++e) {
                A[e] = a0[e];  A[e + 8]  = a1[e];
                B0[e] = b00[e]; B0[e + 8] = b01[e];
                B1[e] = b10[e]; B1[e + 8] = b11[e];
            }
            acc0 = __builtin_amdgcn_wmma_f32_16x16x32_f16(false, A, false, B0,
                                                          (short)0, acc0, false, false);
            acc1 = __builtin_amdgcn_wmma_f32_16x16x32_f16(false, A, false, B1,
                                                          (short)0, acc1, false, false);
        }
        __syncthreads();                         // protect buffer before reuse
    }

    // C/D layout: VGPR r, lanes 0-15 -> M=r, lanes 16-31 -> M=r+8; N = lane&15
    const int crow = bm + wm * 16 + ((lane >> 4) << 3);
    const int ccol = bn + wn * 32 + fr;
#pragma unroll
    for (int r = 0; r < 8; ++r) {
        sim[(size_t)(crow + r) * kN + ccol]      = acc0[r];
        sim[(size_t)(crow + r) * kN + ccol + 16] = acc1[r];
    }
}

// ------------------------------------------------- top-(KNN+1), drop self/top1
__global__ void lame_topk(const float* __restrict__ sim, int* __restrict__ knn) {
    const int i = blockIdx.x, tid = threadIdx.x;
    __shared__ float sv[256];
    __shared__ int   si[256];
    __shared__ int   chs[kKNN + 1];
    int chosen[kKNN + 1];
#pragma unroll 1
    for (int t = 0; t < kKNN + 1; ++t) {
        float bv = -1e30f; int bi = kN;
        for (int j = tid; j < kN; j += 256) {
            bool used = false;
#pragma unroll
            for (int t2 = 0; t2 < kKNN + 1; ++t2)
                if (t2 < t && chosen[t2] == j) used = true;
            if (used) continue;
            const float v = sim[(size_t)i * kN + j];
            if (v > bv || (v == bv && j < bi)) { bv = v; bi = j; }
        }
        sv[tid] = bv; si[tid] = bi; __syncthreads();
        for (int s = 128; s > 0; s >>= 1) {
            if (tid < s) {
                if (sv[tid + s] > sv[tid] ||
                    (sv[tid + s] == sv[tid] && si[tid + s] < si[tid])) {
                    sv[tid] = sv[tid + s]; si[tid] = si[tid + s];
                }
            }
            __syncthreads();
        }
        if (tid == 0) chs[t] = si[0];
        __syncthreads();
        chosen[t] = chs[t];
        __syncthreads();
    }
    if (tid < kKNN) knn[i * kKNN + tid] = chosen[tid + 1];   // drop top-1 (self)
}

// --------------------------------------------- symmetric neighbor lists (CSR-ish)
__global__ void lame_build(const int* __restrict__ knn, int* __restrict__ nbr, int* __restrict__ cnt) {
    const int i = blockIdx.x, tid = threadIdx.x;
    constexpr int CH = kN / 256;              // 8 rows per thread, <=1 match per row
    int loc[CH]; int lc = 0;
    const int j0 = tid * CH;
    for (int j = j0; j < j0 + CH; ++j) {
#pragma unroll
        for (int k = 0; k < kKNN; ++k)
            if (knn[j * kKNN + k] == i) { if (lc < CH) loc[lc++] = j; }
    }
    __shared__ int scan[256];
    scan[tid] = lc; __syncthreads();
    for (int off = 1; off < 256; off <<= 1) {       // inclusive Hillis-Steele scan
        const int v = scan[tid];
        const int add = (tid >= off) ? scan[tid - off] : 0;
        __syncthreads();
        scan[tid] = v + add;
        __syncthreads();
    }
    const int total = scan[255];
    const int base = scan[tid] - lc;                // exclusive prefix (ordered by j)
    const int cap = kMaxDeg - kKNN;
    for (int k = 0; k < lc; ++k) {
        const int pos = base + k;
        if (pos < cap) nbr[i * kMaxDeg + kKNN + pos] = loc[k];
    }
    if (tid == 0) {
#pragma unroll
        for (int k = 0; k < kKNN; ++k) nbr[i * kMaxDeg + k] = knn[i * kKNN + k];
        cnt[i] = kKNN + (total < cap ? total : cap);
    }
}

// ------------------------------------------------- unary = -log(softmax), Y0
__global__ void lame_unary_y0(const float* __restrict__ logits,
                              float* __restrict__ unary, float* __restrict__ Y0) {
    const int i = blockIdx.x, tid = threadIdx.x;
    __shared__ float red[256];
    __shared__ float u[kC];
    const float* lrow = logits + (size_t)i * kC;

    float m = -1e30f;
    for (int c = tid; c < kC; c += 256) m = fmaxf(m, lrow[c]);
    red[tid] = m; __syncthreads();
    for (int t = 128; t > 0; t >>= 1) { if (tid < t) red[tid] = fmaxf(red[tid], red[tid + t]); __syncthreads(); }
    const float m1 = red[0]; __syncthreads();

    float s = 0.f;
    for (int c = tid; c < kC; c += 256) s += __expf(lrow[c] - m1);
    red[tid] = s; __syncthreads();
    for (int t = 128; t > 0; t >>= 1) { if (tid < t) red[tid] += red[tid + t]; __syncthreads(); }
    const float rs1 = 1.f / red[0]; __syncthreads();

    for (int c = tid; c < kC; c += 256) {
        const float p = __expf(lrow[c] - m1) * rs1;
        const float uu = -__logf(p + 1e-10f);
        u[c] = uu;
        unary[(size_t)i * kC + c] = uu;
    }
    __syncthreads();

    // Y0 = softmax(-unary)
    float m2 = -1e30f;
    for (int c = tid; c < kC; c += 256) m2 = fmaxf(m2, -u[c]);
    red[tid] = m2; __syncthreads();
    for (int t = 128; t > 0; t >>= 1) { if (tid < t) red[tid] = fmaxf(red[tid], red[tid + t]); __syncthreads(); }
    const float M2 = red[0]; __syncthreads();
    float s2 = 0.f;
    for (int c = tid; c < kC; c += 256) s2 += __expf(-u[c] - M2);
    red[tid] = s2; __syncthreads();
    for (int t = 128; t > 0; t >>= 1) { if (tid < t) red[tid] += red[tid + t]; __syncthreads(); }
    const float rS2 = 1.f / red[0];
    for (int c = tid; c < kC; c += 256)
        Y0[(size_t)i * kC + c] = __expf(-u[c] - M2) * rS2;
}

// ---------------------------------- one Jacobi step: Ydst = softmax(lam*K@Ysrc - unary)
__global__ void lame_iter(const float* __restrict__ Ysrc, float* __restrict__ Ydst,
                          const float* __restrict__ unary, const int* __restrict__ nbr,
                          const int* __restrict__ cnt, float* __restrict__ e_row,
                          const int* __restrict__ done) {
    const int i = blockIdx.x, tid = threadIdx.x;
    if (*done) {   // converged: propagate result through the ping-pong deterministically
        for (int c = tid; c < kC; c += 256)
            Ydst[(size_t)i * kC + c] = Ysrc[(size_t)i * kC + c];
        return;
    }
    __shared__ float z[kC];
    __shared__ float red[256];
    __shared__ int nb[kMaxDeg];
    const int n = cnt[i];
    if (tid < n) nb[tid] = nbr[i * kMaxDeg + tid];
    __syncthreads();

    float lmax = -1e30f;
    for (int c = tid; c < kC; c += 256) {
        float p = 0.f;
        for (int k = 0; k < n; ++k) p += Ysrc[(size_t)nb[k] * kC + c];
        p *= 0.5f * kLam;                              // pairwise = lam * (K @ Y)
        const float zz = p - unary[(size_t)i * kC + c];
        z[c] = zz;
        lmax = fmaxf(lmax, zz);
    }
    red[tid] = lmax; __syncthreads();
    for (int t = 128; t > 0; t >>= 1) { if (tid < t) red[tid] = fmaxf(red[tid], red[tid + t]); __syncthreads(); }
    const float m = red[0]; __syncthreads();

    float lsum = 0.f;
    for (int c = tid; c < kC; c += 256) lsum += __expf(z[c] - m);
    red[tid] = lsum; __syncthreads();
    for (int t = 128; t > 0; t >>= 1) { if (tid < t) red[tid] += red[tid + t]; __syncthreads(); }
    const float S = red[0];
    const float rS = 1.f / S;
    for (int c = tid; c < kC; c += 256)
        Ydst[(size_t)i * kC + c] = __expf(z[c] - m) * rS;
    // Analytic row energy: unary.Y - pairwise.Y + Y log Y == -(m + log S) for lam=1
    if (tid == 0) e_row[i] = -(m + __logf(S));
}

// --------------------------- deterministic fixed-order energy reduce + done flag
__global__ void lame_energy(const float* __restrict__ e_row, float* oldE, int* done, int iter) {
    if (*done) return;
    __shared__ float red[256];
    const int tid = threadIdx.x;
    float s = 0.f;
    for (int i = tid; i < kN; i += 256) s += e_row[i];
    red[tid] = s; __syncthreads();
    for (int t = 128; t > 0; t >>= 1) { if (tid < t) red[tid] += red[tid + t]; __syncthreads(); }
    if (tid == 0) {
        const float E = red[0];
        const float oe = *oldE;
        if (iter > 1 && fabsf(E - oe) <= 1e-8f * fabsf(oe)) *done = 1;
        *oldE = E;
    }
}

// ------------------------------------------------------------- output copy
__global__ void lame_copy(const float* __restrict__ src, float* __restrict__ dst) {
    const int i = blockIdx.x;
    for (int c = threadIdx.x; c < kC; c += 256)
        dst[(size_t)i * kC + c] = src[(size_t)i * kC + c];
}

extern "C" void kernel_launch(void* const* d_in, const int* in_sizes, int n_in,
                              void* d_out, int out_size, void* d_ws, size_t ws_size,
                              hipStream_t stream) {
    const float* logits = (const float*)d_in[0];   // N x C
    const float* feats  = (const float*)d_in[1];   // N x D
    float* out = (float*)d_out;                    // N x C

    // Workspace carve (all 256B aligned); total ~45.3 MB
    uintptr_t p = (uintptr_t)d_ws;
    auto carve = [&p](size_t bytes) -> void* {
        void* r = (void*)p;
        p += (bytes + 255) & ~(size_t)255;
        return r;
    };
    _Float16* fh   = (_Float16*)carve((size_t)kN * kD * 2);
    float* sim     = (float*)carve((size_t)kN * kN * 4);
    float* unary   = (float*)carve((size_t)kN * kC * 4);
    float* Ya      = (float*)carve((size_t)kN * kC * 4);
    float* Yb      = (float*)carve((size_t)kN * kC * 4);
    int*   knn     = (int*)carve((size_t)kN * kKNN * 4);
    int*   nbr     = (int*)carve((size_t)kN * kMaxDeg * 4);
    int*   cnt     = (int*)carve((size_t)kN * 4);
    float* e_row   = (float*)carve((size_t)kN * 4);
    float* oldE    = (float*)carve(4);
    int*   done    = (int*)carve(4);

    lame_init<<<1, 32, 0, stream>>>(oldE, done);
    lame_normalize<<<kN, 256, 0, stream>>>(feats, fh);
    lame_gram_wmma<<<(kN / 64) * (kN / 64), 256, 0, stream>>>(fh, sim);
    lame_topk<<<kN, 256, 0, stream>>>(sim, knn);
    lame_build<<<kN, 256, 0, stream>>>(knn, nbr, cnt);
    lame_unary_y0<<<kN, 256, 0, stream>>>(logits, unary, Ya);

    for (int it = 0; it < kSteps; ++it) {
        const float* src = (it & 1) ? Yb : Ya;
        float*       dst = (it & 1) ? Ya : Yb;
        lame_iter<<<kN, 256, 0, stream>>>(src, dst, unary, nbr, cnt, e_row, done);
        lame_energy<<<1, 256, 0, stream>>>(e_row, oldE, done, it);
    }
    // kSteps = 100 (even) -> last write (it=99, odd) landed in Ya
    lame_copy<<<kN, 256, 0, stream>>>(Ya, out);
    (void)in_sizes; (void)n_in; (void)out_size; (void)ws_size;
}
